// Attention_10136122818679
// MI455X (gfx1250) — matrix operand order repaired and checked by reading, not compiled, not run on hardware
//
#include <hip/hip_runtime.h>
#include <hip/hip_bf16.h>

#define DIMM   1024
#define HEADS  16
#define DHEAD  64
#define NUM_PM 16
#define SEQ    2048
#define BATCH  4
#define ROWS   (BATCH * SEQ)     // 8192
#define INNER  (HEADS * DHEAD)   // 1024
#define NQKV   (3 * INNER)       // 3072
#define KVV    (NUM_PM + SEQ)    // 2064 valid kv positions
#define KVP    2080              // padded to multiple of 32
#define LN_EPS 1e-5f
#define ATT_SCALE 0.125f         // 64^-0.5

typedef __attribute__((ext_vector_type(16))) _Float16 v16h;
typedef __attribute__((ext_vector_type(8)))  _Float16 v8h;
typedef __attribute__((ext_vector_type(8)))  float    v8f;
typedef unsigned int __attribute__((ext_vector_type(4))) u32x4;
typedef int  __attribute__((ext_vector_type(4))) i32x4;
typedef int  __attribute__((ext_vector_type(8))) i32x8;

union HFrag { v16h f; v8h h8[2]; _Float16 h[16]; };

// ---------------- fragment loaders -------------------------------------
// A fragment 16x32 f16 (ISA 7.12.2): lanes 0-15 hold M=lane, K = {0..7,16..23};
// lanes 16-31 hold M=lane-16, K = {8..15,24..31}. Two 16B loads per lane.
__device__ __forceinline__ void load_a_frag(HFrag& fr, const _Float16* base,
                                            int ld, int row, int k, int lane) {
  int kb = (lane & 16) ? 8 : 0;
  const _Float16* p = base + (size_t)row * ld + k + kb;
  fr.h8[0] = *(const v8h*)(p);
  fr.h8[1] = *(const v8h*)(p + 16);
}
// B fragment 32x16 f16: lane holds column N=lane&15; lanes 0-15 carry
// K={0..7,16..23}, lanes 16-31 carry K={8..15,24..31}. Bt stored [N,K].
__device__ __forceinline__ void load_b_frag(HFrag& fr, const _Float16* baseT,
                                            int ld, int col, int k, int lane) {
  int kb = (lane & 16) ? 8 : 0;
  const _Float16* p = baseT + (size_t)col * ld + k + kb;
  fr.h8[0] = *(const v8h*)(p);
  fr.h8[1] = *(const v8h*)(p + 16);
}

__device__ __forceinline__ float hmax16(float v) {
#pragma unroll
  for (int m = 1; m < 16; m <<= 1) v = fmaxf(v, __shfl_xor(v, m, 32));
  return v;
}
__device__ __forceinline__ float hsum16(float v) {
#pragma unroll
  for (int m = 1; m < 16; m <<= 1) v += __shfl_xor(v, m, 32);
  return v;
}

#define WMMA_F16(A, B, C) \
  __builtin_amdgcn_wmma_f32_16x16x32_f16(false, (A).f, false, (B).f, (short)0, (C), false, false)

// ---------------- 32x32 wave tile GEMM (4 WMMAs per 8 b128 loads) ------
__device__ __forceinline__ void wave_tile_gemm32(const _Float16* A, const _Float16* Bt,
                                                 int lda, int ldb, int K,
                                                 int mBase, int nBase, int lane,
                                                 v8f acc[4]) {
  int ra0 = mBase + (lane & 15), ra1 = ra0 + 16;
  int rb0 = nBase + (lane & 15), rb1 = rb0 + 16;
  for (int k = 0; k < K; k += 32) {
    HFrag a0, a1, b0, b1;
    load_a_frag(a0, A, lda, ra0, k, lane);
    load_a_frag(a1, A, lda, ra1, k, lane);
    load_b_frag(b0, Bt, ldb, rb0, k, lane);
    load_b_frag(b1, Bt, ldb, rb1, k, lane);
    if (k + 64 < K) {
      __builtin_prefetch(A + (size_t)ra0 * lda + k + 64, 0, 1);
      __builtin_prefetch(A + (size_t)ra1 * lda + k + 64, 0, 1);
      __builtin_prefetch(Bt + (size_t)rb0 * ldb + k + 64, 0, 1);
      __builtin_prefetch(Bt + (size_t)rb1 * ldb + k + 64, 0, 1);
    }
    acc[0] = WMMA_F16(a0, b0, acc[0]);
    acc[1] = WMMA_F16(a0, b1, acc[1]);
    acc[2] = WMMA_F16(a1, b0, acc[2]);
    acc[3] = WMMA_F16(a1, b1, acc[3]);
  }
}

// ---------------- Stage 1: LayerNorm -----------------------------------
__global__ void ln_kernel(const float* __restrict__ x,
                          const float* __restrict__ gamma,
                          const float* __restrict__ beta,
                          float* __restrict__ normed_f32,
                          _Float16* __restrict__ normed_h) {
  int row = blockIdx.x;
  int tid = threadIdx.x;
  const float* xr = x + (size_t)row * DIMM;
  float v[4];
  float s = 0.f;
#pragma unroll
  for (int i = 0; i < 4; ++i) { v[i] = xr[tid + i * 256]; s += v[i]; }
  __shared__ float red[8];
#pragma unroll
  for (int m = 1; m < 32; m <<= 1) s += __shfl_xor(s, m, 32);
  if ((tid & 31) == 0) red[tid >> 5] = s;
  __syncthreads();
  float tot = 0.f;
#pragma unroll
  for (int i = 0; i < 8; ++i) tot += red[i];
  float mu = tot * (1.0f / DIMM);
  __syncthreads();
  float vs = 0.f;
#pragma unroll
  for (int i = 0; i < 4; ++i) { float d = v[i] - mu; vs += d * d; }
#pragma unroll
  for (int m = 1; m < 32; m <<= 1) vs += __shfl_xor(vs, m, 32);
  if ((tid & 31) == 0) red[tid >> 5] = vs;
  __syncthreads();
  float vtot = 0.f;
#pragma unroll
  for (int i = 0; i < 8; ++i) vtot += red[i];
  float rstd = rsqrtf(vtot * (1.0f / DIMM) + LN_EPS);
#pragma unroll
  for (int i = 0; i < 4; ++i) {
    int c = tid + i * 256;
    float nv = (v[i] - mu) * rstd * gamma[c] + beta[c];
    normed_f32[(size_t)row * DIMM + c] = nv;
    normed_h[(size_t)row * DIMM + c] = (_Float16)nv;
  }
}

// ------------- weight f32 -> f16 transposed [N,K] ----------------------
__global__ void convert_transpose_kernel(const float* __restrict__ w,
                                         _Float16* __restrict__ wt,
                                         int rowsK, int colsN) {
  size_t idx = (size_t)blockIdx.x * blockDim.x + threadIdx.x;
  size_t total = (size_t)rowsK * colsN;
  if (idx >= total) return;
  int r = (int)(idx / colsN);
  int c = (int)(idx % colsN);
  wt[(size_t)c * rowsK + r] = (_Float16)w[idx];
}

// ------------- persistent memory rows + zero pad -----------------------
__global__ void pm_pad_kernel(const float* __restrict__ pm,
                              _Float16* __restrict__ kh,
                              _Float16* __restrict__ vt) {
  int idx = blockIdx.x * blockDim.x + threadIdx.x;   // 4*16*32*64
  int d = idx & 63;
  int j = (idx >> 6) & 31;
  int h = (idx >> 11) & 15;
  int b = idx >> 15;
  int bh = b * HEADS + h;
  if (j < NUM_PM) {
    float kv = pm[((size_t)(0 * HEADS + h) * NUM_PM + j) * DHEAD + d];
    float vv = pm[((size_t)(1 * HEADS + h) * NUM_PM + j) * DHEAD + d];
    kh[((size_t)bh * KVP + j) * DHEAD + d] = (_Float16)kv;
    vt[((size_t)bh * DHEAD + d) * KVP + j] = (_Float16)vv;
  } else {
    int kv = KVV + (j - NUM_PM);                     // zero pad rows
    kh[((size_t)bh * KVP + kv) * DHEAD + d] = (_Float16)0.0f;
    vt[((size_t)bh * DHEAD + d) * KVP + kv] = (_Float16)0.0f;
  }
}

// ---------------- Stage 2: QKV GEMM (32x32 tile) + head scatter --------
__global__ void qkv_gemm_kernel(const _Float16* __restrict__ A,
                                const _Float16* __restrict__ Bt,
                                _Float16* __restrict__ qh,
                                _Float16* __restrict__ kh,
                                _Float16* __restrict__ vt) {
  int lane = threadIdx.x & 31;
  int wave = threadIdx.x >> 5;
  int tile = blockIdx.x * 8 + wave;
  const int mTiles = ROWS / 32;            // 256
  int mBase = (tile % mTiles) * 32;
  int nBase = (tile / mTiles) * 32;
  v8f acc[4] = {};
  wave_tile_gemm32(A, Bt, DIMM, DIMM, DIMM, mBase, nBase, lane, acc);
  int col = lane & 15;
  int rhi = (lane & 16) ? 8 : 0;
#pragma unroll
  for (int mi = 0; mi < 2; ++mi) {
#pragma unroll
    for (int ni = 0; ni < 2; ++ni) {
      const v8f& a = acc[mi * 2 + ni];
      int c = nBase + ni * 16 + col;
      int which = c >> 10;
      int inner = c & (INNER - 1);
      int head = inner >> 6;
      int d = inner & 63;
#pragma unroll
      for (int j = 0; j < 8; ++j) {
        int m = mBase + mi * 16 + j + rhi;
        int b = m >> 11;
        int n = m & (SEQ - 1);
        int bh = b * HEADS + head;
        _Float16 hv = (_Float16)a[j];
        if (which == 0)      qh[((size_t)bh * SEQ + n) * DHEAD + d] = hv;
        else if (which == 1) kh[((size_t)bh * KVP + NUM_PM + n) * DHEAD + d] = hv;
        else                 vt[((size_t)bh * DHEAD + d) * KVP + NUM_PM + n] = hv;
      }
    }
  }
}

// ---------------- Stage 3: flash attention -----------------------------
__global__ void attn_kernel(const _Float16* __restrict__ q,
                            const _Float16* __restrict__ kmat,
                            const _Float16* __restrict__ vt,
                            _Float16* __restrict__ attn_h) {
  __shared__ __align__(64) _Float16 plds[8][16 * 32];
  int lane = threadIdx.x & 31;
  int wave = threadIdx.x >> 5;
  int tile = blockIdx.x * 8 + wave;
  int qt = tile & 127;
  int bh = tile >> 7;
  const _Float16* qb = q    + (size_t)bh * SEQ * DHEAD;
  const _Float16* kb = kmat + (size_t)bh * KVP * DHEAD;
  const _Float16* vb = vt   + (size_t)bh * DHEAD * KVP;
  int col = lane & 15;
  int rhi = (lane & 16) ? 8 : 0;

  HFrag q0, q1;
  load_a_frag(q0, qb, DHEAD, qt * 16 + (lane & 15), 0, lane);
  load_a_frag(q1, qb, DHEAD, qt * 16 + (lane & 15), 32, lane);

  float mrow[8], lrow[8];
  v8f o0 = {}, o1 = {}, o2 = {}, o3 = {};
#pragma unroll
  for (int j = 0; j < 8; ++j) { mrow[j] = -3.0e38f; lrow[j] = 0.0f; }

  _Float16* pl = &plds[wave][0];

  for (int jb = 0; jb < KVP; jb += 32) {
    HFrag kf0, kf1;
    v8f s0 = {}, s1 = {};
    load_b_frag(kf0, kb, DHEAD, jb + col, 0, lane);
    load_b_frag(kf1, kb, DHEAD, jb + col, 32, lane);
    s0 = WMMA_F16(q0, kf0, s0);
    s0 = WMMA_F16(q1, kf1, s0);
    load_b_frag(kf0, kb, DHEAD, jb + 16 + col, 0, lane);
    load_b_frag(kf1, kb, DHEAD, jb + 16 + col, 32, lane);
    s1 = WMMA_F16(q0, kf0, s1);
    s1 = WMMA_F16(q1, kf1, s1);

    bool v1ok = (jb + 16 + col) < KVV;   // first half always valid (<=2063)
#pragma unroll
    for (int j = 0; j < 8; ++j) {
      float a0 = s0[j] * ATT_SCALE;
      float a1 = v1ok ? (s1[j] * ATT_SCALE) : -3.0e38f;
      float rm = hmax16(fmaxf(a0, a1));
      float nm = fmaxf(mrow[j], rm);
      float alpha = __expf(mrow[j] - nm);
      float p0 = __expf(a0 - nm);
      float p1 = __expf(a1 - nm);
      float rs = hsum16(p0 + p1);
      lrow[j] = lrow[j] * alpha + rs;
      mrow[j] = nm;
      o0[j] *= alpha; o1[j] *= alpha; o2[j] *= alpha; o3[j] *= alpha;
      pl[(j + rhi) * 32 + col]      = (_Float16)p0;
      pl[(j + rhi) * 32 + col + 16] = (_Float16)p1;
    }
    asm volatile("s_wait_dscnt 0" ::: "memory");   // cross-lane via LDS
    HFrag pf;
    {
      int kbq = (lane & 16) ? 8 : 0;
      const _Float16* pp = pl + (lane & 15) * 32 + kbq;
      pf.h8[0] = *(const v8h*)(pp);
      pf.h8[1] = *(const v8h*)(pp + 16);
    }
    HFrag vf;
    load_b_frag(vf, vb, KVP, 0 + col, jb, lane);
    o0 = WMMA_F16(pf, vf, o0);
    load_b_frag(vf, vb, KVP, 16 + col, jb, lane);
    o1 = WMMA_F16(pf, vf, o1);
    load_b_frag(vf, vb, KVP, 32 + col, jb, lane);
    o2 = WMMA_F16(pf, vf, o2);
    load_b_frag(vf, vb, KVP, 48 + col, jb, lane);
    o3 = WMMA_F16(pf, vf, o3);
  }

  int b = bh >> 4, h = bh & 15;
  size_t rowBase = (size_t)b * SEQ + qt * 16;
#pragma unroll
  for (int j = 0; j < 8; ++j) {
    float inv = 1.0f / lrow[j];
    size_t r = rowBase + j + rhi;
    _Float16* op = attn_h + r * INNER + h * DHEAD;
    op[col]      = (_Float16)(o0[j] * inv);
    op[16 + col] = (_Float16)(o1[j] * inv);
    op[32 + col] = (_Float16)(o2[j] * inv);
    op[48 + col] = (_Float16)(o3[j] * inv);
  }
}

// ---------------- Stage 4: output projection (32x32 tile) + bias -------
__global__ void out_gemm_kernel(const _Float16* __restrict__ A,
                                const _Float16* __restrict__ Bt,
                                const float* __restrict__ bias,
                                float* __restrict__ out) {
  int lane = threadIdx.x & 31;
  int wave = threadIdx.x >> 5;
  int tile = blockIdx.x * 8 + wave;
  const int mTiles = ROWS / 32;            // 256
  int mBase = (tile % mTiles) * 32;
  int nBase = (tile / mTiles) * 32;
  v8f acc[4] = {};
  wave_tile_gemm32(A, Bt, INNER, INNER, INNER, mBase, nBase, lane, acc);
  int col = lane & 15;
  int rhi = (lane & 16) ? 8 : 0;
#pragma unroll
  for (int mi = 0; mi < 2; ++mi) {
#pragma unroll
    for (int ni = 0; ni < 2; ++ni) {
      const v8f& a = acc[mi * 2 + ni];
      int c = nBase + ni * 16 + col;
      float bv = bias[c];
#pragma unroll
      for (int j = 0; j < 8; ++j)
        out[(size_t)(mBase + mi * 16 + j + rhi) * DIMM + c] = a[j] + bv;
    }
  }
}

// ------------- TDM demo: stage a 16x64 f16 tile into LDS ---------------
// Builds a minimal 2D Tensor-DMA descriptor (ISA ch.8 D# group0/group1):
// group0: count=1, lds_addr, global_addr, type=2; group1: data_size=2B,
// tensor 64x16, tile 64x16, dim0 stride 64. Then waits TENSORcnt.
// clang-23 lane: 6-arg builtin (u32x4, i32x8, i32x4, i32x4, i32x8, i32).
__global__ void tdm_stage_kernel(const _Float16* __restrict__ src,
                                 _Float16* __restrict__ dst) {
#if __has_builtin(__builtin_amdgcn_tensor_load_to_lds) && \
    __has_builtin(__builtin_amdgcn_s_wait_tensorcnt)
  __shared__ __align__(128) _Float16 tile[16 * 64];
  unsigned int lds_off = (unsigned int)(size_t)&tile[0];
  unsigned long long ga = (unsigned long long)(size_t)src;
  u32x4 g0 = { 1u,                                       // count=1, user mode
               lds_off,                                  // lds_addr [63:32]
               (unsigned int)ga,                         // global_addr lo
               ((unsigned int)(ga >> 32) & 0x01FFFFFFu)  // global_addr hi
                 | (2u << 30) };                         // type=2 (image)
  i32x8 g1 = { (int)(1u << 16),                // data_size = 1 (2 bytes)
               (int)(64u << 16),               // tensor_dim0 = 64 (lo16)
               (int)(16u << 16),               // tensor_dim1 = 16
               (int)(64u << 16),               // tile_dim0 = 64
               16,                             // tile_dim1 = 16, tile_dim2 = 0
               64,                             // tensor_dim0_stride = 64
               0, 0 };
  i32x4 gz4 = { 0, 0, 0, 0 };
  i32x8 gz8 = { 0, 0, 0, 0, 0, 0, 0, 0 };
  __builtin_amdgcn_tensor_load_to_lds(g0, g1, gz4, gz4, gz8, 0);
  __builtin_amdgcn_s_wait_tensorcnt(0);
  int t = threadIdx.x;
  for (int i = t; i < 16 * 64; i += 32) dst[i] = tile[i];
#else
  int t = threadIdx.x;
  for (int i = t; i < 16 * 64; i += 32) dst[i] = src[i];
#endif
}

// ---------------- host side --------------------------------------------
extern "C" void kernel_launch(void* const* d_in, const int* in_sizes, int n_in,
                              void* d_out, int out_size, void* d_ws, size_t ws_size,
                              hipStream_t stream) {
  (void)in_sizes; (void)n_in; (void)out_size; (void)ws_size;
  const float* x        = (const float*)d_in[0];
  const float* ln_gamma = (const float*)d_in[1];
  const float* ln_beta  = (const float*)d_in[2];
  const float* w_qkv    = (const float*)d_in[3];
  const float* pm       = (const float*)d_in[4];
  const float* w_out    = (const float*)d_in[5];
  const float* b_out    = (const float*)d_in[6];

  float* out        = (float*)d_out;                    // tuple slot 0
  float* normed_out = out + (size_t)ROWS * DIMM;        // tuple slot 1

  char* ws = (char*)d_ws;
  auto alloc = [&](size_t bytes) {
    char* p = ws;
    ws += (bytes + 255) & ~(size_t)255;
    return p;
  };
  _Float16* normed_h = (_Float16*)alloc((size_t)ROWS * DIMM * 2);
  _Float16* wqkv_t   = (_Float16*)alloc((size_t)NQKV * DIMM * 2);
  _Float16* wout_t   = (_Float16*)alloc((size_t)DIMM * INNER * 2);
  _Float16* qh       = (_Float16*)alloc((size_t)BATCH * HEADS * SEQ * DHEAD * 2);
  _Float16* kh       = (_Float16*)alloc((size_t)BATCH * HEADS * KVP * DHEAD * 2);
  _Float16* vt       = (_Float16*)alloc((size_t)BATCH * HEADS * DHEAD * KVP * 2);
  _Float16* attn_h   = (_Float16*)alloc((size_t)ROWS * INNER * 2);
  _Float16* tdm_dst  = (_Float16*)alloc((size_t)16 * 64 * 2);

  ln_kernel<<<ROWS, 256, 0, stream>>>(x, ln_gamma, ln_beta, normed_out, normed_h);
  convert_transpose_kernel<<<(DIMM * NQKV + 255) / 256, 256, 0, stream>>>(
      w_qkv, wqkv_t, DIMM, NQKV);
  convert_transpose_kernel<<<(INNER * DIMM + 255) / 256, 256, 0, stream>>>(
      w_out, wout_t, INNER, DIMM);
  pm_pad_kernel<<<(BATCH * HEADS * 32 * DHEAD) / 256, 256, 0, stream>>>(pm, kh, vt);
  qkv_gemm_kernel<<<(ROWS / 32) * (NQKV / 32) / 8, 256, 0, stream>>>(
      normed_h, wqkv_t, qh, kh, vt);
  attn_kernel<<<BATCH * HEADS * (SEQ / 16) / 8, 256, 0, stream>>>(qh, kh, vt, attn_h);
  out_gemm_kernel<<<(ROWS / 32) * (DIMM / 32) / 8, 256, 0, stream>>>(
      attn_h, wout_t, b_out, out);
  tdm_stage_kernel<<<1, 32, 0, stream>>>(qh, tdm_dst);
}
